// modelf_68272800137614
// MI455X (gfx1250) — compile-verified, matrix operand order checked by reading, not run on hardware
//
#include <hip/hip_runtime.h>
#include <hip/hip_bf16.h>
#include <stddef.h>

// ---------------- model constants ----------------
constexpr int TX = 30, TY = 10, NA = 32, NS = 64, HV = 128, MV = 64;
constexpr int BM = 16;          // batch rows per workgroup (WMMA M)
constexpr int NTH = 256;        // 8 waves of 32

// LDS row strides (elements), padded to de-phase banks
constexpr int XLDH = 136;       // X tile f16 row stride (>=128)
constexpr int ALDH = 72;        // encoder output a tile f16 row stride (>=64)
constexpr int HLDH = 40;        // encoder h f16 row stride (>=32)
constexpr int SLDH = 72;        // decoder s f16 row stride (>=64)
constexpr int CLDH = 72;        // context f16 row stride (>=64)
constexpr int ZLDE = 136;       // encoder z f32 row stride (>=128)
constexpr int ZLDD = 264;       // decoder z f32 row stride (>=256)
constexpr int ELD  = 16;        // e buffer row stride f32

// LDS carve (bytes, all 16B aligned)
constexpr int OFF_XH  = 0;                             // 30*16*136*2 = 130560
constexpr int OFF_AT  = OFF_XH  + TX*BM*XLDH*2;        // +69120
constexpr int OFF_HF  = OFF_AT  + TX*BM*ALDH*2;        // +1280
constexpr int OFF_HR  = OFF_HF  + BM*HLDH*2;           // +1280
constexpr int OFF_S16 = OFF_HR  + BM*HLDH*2;           // +2304
constexpr int OFF_CTX = OFF_S16 + BM*SLDH*2;           // +2304
constexpr int OFF_Z   = OFF_CTX + BM*CLDH*2;           // z union: enc 2*16*136 f32, dec 16*264 f32
constexpr int ZF32    = 2*BM*ZLDE;                     // 4352 floats (>= 16*264)
constexpr int OFF_EB  = OFF_Z   + ZF32*4;              // e buffer 30*16*16 f32
constexpr int OFF_EN  = OFF_EB  + TX*BM*ELD*4;         // en/alphas 16*32 f32
constexpr int OFF_BIA = OFF_EN  + BM*32*4;             // biases: 612 floats
constexpr int SMEM_BYTES = OFF_BIA + 612*4;            // ~259.4 KB (< 320 KB WGP LDS)

typedef __attribute__((ext_vector_type(16))) _Float16 v16h;
typedef __attribute__((ext_vector_type(8)))  float    v8f;
typedef __attribute__((ext_vector_type(4)))  float    f4;
typedef __attribute__((ext_vector_type(4)))  _Float16 h4;

static __device__ __forceinline__ v8f wmmaf(v16h a, v16h b, v8f c) {
  return __builtin_amdgcn_wmma_f32_16x16x32_f16(false, a, false, b, (short)0, c, false, false);
}

// Branchless fast exp: v_exp_f32 computes 2^x
static __device__ __forceinline__ float fexp(float x) {
  return __builtin_amdgcn_exp2f(x * 1.4426950408889634f);
}
// Fast tanh: native V_TANH_F32 on CDNA5 if the builtin is declared, else a
// branchless exp2/rcp form (exact at saturation: x>>0 -> 1, x<<0 -> -1).
static __device__ __forceinline__ float ftanh(float x) {
#if __has_builtin(__builtin_amdgcn_tanhf)
  return __builtin_amdgcn_tanhf(x);
#elif __has_builtin(__builtin_amdgcn_tanh_f32)
  return __builtin_amdgcn_tanh_f32(x);
#else
  float e = __builtin_amdgcn_exp2f(x * 2.885390081777927f);  // exp(2x)
  return 1.0f - 2.0f * __builtin_amdgcn_rcpf(e + 1.0f);
#endif
}
static __device__ __forceinline__ float sigf(float x) {
  return 0.5f + 0.5f * ftanh(0.5f * x);
}

// A fragment (16x32 f16, M=batch rows) from LDS; p points at [row0, k-chunk base]
static __device__ __forceinline__ v16h ldsA(const _Float16* p, int stride, int lane) {
  const int r = lane & 15;
  const int g = (lane >> 4) << 3;  // 0 or 8
  const _Float16* q = p + r * stride + g;
  v16h v;
#pragma unroll
  for (int i = 0; i < 8; ++i) { v[i] = q[i]; v[8 + i] = q[16 + i]; }
  return v;
}

// B fragment (32x16 f16, K x N): B[k][n] = W[n0+n][kb+k], W row-major [outDim, ldk] f32.
// Lanes with n >= nvalid supply zeros (used to pad N=10 attention GEMM to 16).
static __device__ __forceinline__ v16h bfrag(const float* W, int ldk, int n0, int kb,
                                             int lane, int nvalid) {
  const int n = lane & 15;
  const int g = (lane >> 4) << 3;
  v16h v = {};
  if (n < nvalid) {
    const float* q = W + (size_t)(n0 + n) * ldk + kb + g;
#pragma unroll
    for (int i = 0; i < 8; ++i) { v[i] = (_Float16)q[i]; v[8 + i] = (_Float16)q[16 + i]; }
  }
  return v;
}

// D (16x16 f32) scatter to LDS f32 buffer: lane l, vgpr v -> row v+8*(l>>4), col n0+(l&15)
static __device__ __forceinline__ void storeD(float* z, int stride, int n0, int lane, v8f acc) {
  const int n  = lane & 15;
  const int mb = (lane >> 4) << 3;
#pragma unroll
  for (int v = 0; v < 8; ++v) z[(v + mb) * stride + n0 + n] = acc[v];
}

__global__ void __launch_bounds__(NTH)
nmt_attn_fused(const float* __restrict__ X,
               const float* __restrict__ W_ih_f, const float* __restrict__ W_hh_f, const float* __restrict__ b_f,
               const float* __restrict__ W_ih_r, const float* __restrict__ W_hh_r, const float* __restrict__ b_r,
               const float* __restrict__ W1, const float* __restrict__ b1,
               const float* __restrict__ W2, const float* __restrict__ b2,
               const float* __restrict__ W_ih_p, const float* __restrict__ W_hh_p, const float* __restrict__ b_p,
               const float* __restrict__ Wo, const float* __restrict__ bo,
               float* __restrict__ out, int Bn) {
  extern __shared__ char smem[];
  _Float16* Xh   = (_Float16*)(smem + OFF_XH);
  _Float16* aT   = (_Float16*)(smem + OFF_AT);
  _Float16* hf16 = (_Float16*)(smem + OFF_HF);
  _Float16* hr16 = (_Float16*)(smem + OFF_HR);
  _Float16* s16  = (_Float16*)(smem + OFF_S16);
  _Float16* ctx16= (_Float16*)(smem + OFF_CTX);
  float*    zbuf = (float*)   (smem + OFF_Z);
  float*    e_buf= (float*)   (smem + OFF_EB);
  float*    en   = (float*)   (smem + OFF_EN);
  float*    bias = (float*)   (smem + OFF_BIA);
  float* bF  = bias;        float* bR  = bias + 128;  float* bP  = bias + 256;
  float* b1p = bias + 512;  float* W2p = bias + 528;  float* boS = bias + 544;
  float* b2s = bias + 608;

  const int tid  = threadIdx.x;
  const int lane = tid & 31;
  const int wv   = tid >> 5;
  const int b0   = blockIdx.x * BM;

  // ---- stage biases ----
  for (int i = tid; i < 128; i += NTH) { bF[i] = b_f[i]; bR[i] = b_r[i]; }
  for (int i = tid; i < 256; i += NTH) { bP[i] = b_p[i]; }
  if (tid < 16) { b1p[tid] = (tid < 10) ? b1[tid] : 0.0f; W2p[tid] = (tid < 10) ? W2[tid] : 0.0f; }
  if (tid < 64) { boS[tid] = bo[tid]; }
  if (tid == 0) { b2s[0] = b2[0]; }
  // ---- zero recurrent f16 states ----
  for (int i = tid; i < BM * HLDH; i += NTH) { hf16[i] = (_Float16)0.0f; hr16[i] = (_Float16)0.0f; }
  for (int i = tid; i < BM * SLDH; i += NTH) { s16[i] = (_Float16)0.0f; }

  // ---- stage X tile -> LDS f16, vectorized float4 -> half4 (read X exactly once) ----
  {
    const f4* X4 = (const f4*)X;
    constexpr int ROW4 = TX * HV / 4;  // 960 float4 per batch row
    for (int i = tid; i < BM * ROW4; i += NTH) {
      const int r = i / ROW4;
      const int rem = i - r * ROW4;            // float4 index within row
      const int t = rem / (HV / 4);
      const int k = (rem - t * (HV / 4)) * 4;  // element index, multiple of 4
      f4 x = {0.0f, 0.0f, 0.0f, 0.0f};
      if (b0 + r < Bn) x = X4[(size_t)(b0 + r) * ROW4 + rem];
      h4 hh;
#pragma unroll
      for (int q = 0; q < 4; ++q) hh[q] = (_Float16)x[q];
      *(h4*)(&Xh[(t * BM + r) * XLDH + k]) = hh;  // 8B-aligned LDS store
    }
  }

  // ---- encoder weight fragments (registers, per wave: gate cols [16w,16w+16)) ----
  const int n0e = 16 * wv;
  v16h BfI[4], BrI[4], BfH, BrH;
#pragma unroll
  for (int c = 0; c < 4; ++c) {
    BfI[c] = bfrag(W_ih_f, HV, n0e, 32 * c, lane, 16);
    BrI[c] = bfrag(W_ih_r, HV, n0e, 32 * c, lane, 16);
  }
  BfH = bfrag(W_hh_f, NA, n0e, 0, lane, 16);
  BrH = bfrag(W_hh_r, NA, n0e, 0, lane, 16);

  float cFr[2] = {0.0f, 0.0f}, cRr[2] = {0.0f, 0.0f};
  float* zF = zbuf;
  float* zR = zbuf + BM * ZLDE;

  __syncthreads();

  // ======================= encoder: 30 bidirectional LSTM steps =======================
  for (int t = 0; t < TX; ++t) {
    const int tr = TX - 1 - t;
    // gates z = x@W_ih^T + h@W_hh^T  (bias added in scalar pass)
    v8f aF = {}, aR = {};
#pragma unroll
    for (int c = 0; c < 4; ++c) {
      aF = wmmaf(ldsA(Xh + (t  * BM) * XLDH + 32 * c, XLDH, lane), BfI[c], aF);
      aR = wmmaf(ldsA(Xh + (tr * BM) * XLDH + 32 * c, XLDH, lane), BrI[c], aR);
    }
    aF = wmmaf(ldsA(hf16, HLDH, lane), BfH, aF);
    aR = wmmaf(ldsA(hr16, HLDH, lane), BrH, aR);
    storeD(zF, ZLDE, n0e, lane, aF);
    storeD(zR, ZLDE, n0e, lane, aR);
    __syncthreads();

    // scalar gate pass: thread -> (row, 2 hidden units) per direction
    {
      const int r = tid >> 4, j0 = (tid & 15) * 2;
#pragma unroll
      for (int k = 0; k < 2; ++k) {
        const int j = j0 + k;
        // forward
        float zi = zF[r * ZLDE + j]      + bF[j];
        float zf = zF[r * ZLDE + 32 + j] + bF[32 + j];
        float zg = zF[r * ZLDE + 64 + j] + bF[64 + j];
        float zo = zF[r * ZLDE + 96 + j] + bF[96 + j];
        cFr[k] = sigf(zf) * cFr[k] + sigf(zi) * ftanh(zg);
        float h = sigf(zo) * ftanh(cFr[k]);
        hf16[r * HLDH + j] = (_Float16)h;
        aT[(t * BM + r) * ALDH + j] = (_Float16)h;          // a_f at time t
        // reverse (state after consuming x_{29..29-t} -> a_r index 29-t)
        zi = zR[r * ZLDE + j]      + bR[j];
        zf = zR[r * ZLDE + 32 + j] + bR[32 + j];
        zg = zR[r * ZLDE + 64 + j] + bR[64 + j];
        zo = zR[r * ZLDE + 96 + j] + bR[96 + j];
        cRr[k] = sigf(zf) * cRr[k] + sigf(zi) * ftanh(zg);
        h = sigf(zo) * ftanh(cRr[k]);
        hr16[r * HLDH + j] = (_Float16)h;
        aT[(tr * BM + r) * ALDH + 32 + j] = (_Float16)h;    // a_r at time 29-t
      }
    }
    __syncthreads();
  }

  // ---- decoder weight fragments ----
  v16h B1f[4];
#pragma unroll
  for (int c = 0; c < 4; ++c) B1f[c] = bfrag(W1, 2 * NA + NS, 0, 32 * c, lane, 10);
  v16h Bip[2][2], Bhp[2][2];
#pragma unroll
  for (int jj = 0; jj < 2; ++jj)
#pragma unroll
    for (int c = 0; c < 2; ++c) {
      const int n0 = 32 * wv + 16 * jj;
      Bip[jj][c] = bfrag(W_ih_p, 2 * NA, n0, 32 * c, lane, 16);
      Bhp[jj][c] = bfrag(W_hh_p, NS,     n0, 32 * c, lane, 16);
    }
  v16h BoF[2] = {{}, {}};
  if (wv < 4) {
#pragma unroll
    for (int c = 0; c < 2; ++c) BoF[c] = bfrag(Wo, NS, 16 * wv, 32 * c, lane, 16);
  }
  float cD[4] = {0.0f, 0.0f, 0.0f, 0.0f};
  float* z2 = zbuf;  // reuse encoder z buffer: [16][ZLDD]

  // ======================= decoder: 10 attention + LSTM steps =======================
  for (int ty = 0; ty < TY; ++ty) {
    // --- attention scores: e = tanh([a_t | s] @ W1^T + b1), K=128, N=10 (padded 16) ---
    for (int t = wv; t < TX; t += 8) {            // wave-uniform trip counts
      v8f acc = {};
      acc = wmmaf(ldsA(aT + (t * BM) * ALDH + 0,  ALDH, lane), B1f[0], acc);
      acc = wmmaf(ldsA(aT + (t * BM) * ALDH + 32, ALDH, lane), B1f[1], acc);
      acc = wmmaf(ldsA(s16 + 0,  SLDH, lane), B1f[2], acc);
      acc = wmmaf(ldsA(s16 + 32, SLDH, lane), B1f[3], acc);
      const int n = lane & 15, mb = (lane >> 4) << 3;
#pragma unroll
      for (int v = 0; v < 8; ++v)
        e_buf[(t * BM + (v + mb)) * ELD + n] = ftanh(acc[v] + b1p[n]);
    }
    __syncthreads();

    // --- en = relu(e @ W2^T + b2): 480 scalars ---
#pragma unroll
    for (int q = 0; q < 2; ++q) {
      const int ii = 2 * tid + q;
      if (ii < BM * TX) {
        const int r = ii / TX, t = ii - r * TX;
        float v = b2s[0];
#pragma unroll
        for (int j = 0; j < 10; ++j) v += e_buf[(t * BM + r) * ELD + j] * W2p[j];
        en[r * 32 + t] = fmaxf(v, 0.0f);
      }
    }
    __syncthreads();

    // --- softmax over Tx per batch row ---
    if (tid < BM) {
      const int r = tid;
      float m = -1e30f;
      for (int t = 0; t < TX; ++t) m = fmaxf(m, en[r * 32 + t]);
      float s = 0.0f;
      for (int t = 0; t < TX; ++t) { float e = fexp(en[r * 32 + t] - m); en[r * 32 + t] = e; s += e; }
      const float inv = __builtin_amdgcn_rcpf(s);
      for (int t = 0; t < TX; ++t) en[r * 32 + t] *= inv;
    }
    __syncthreads();

    // --- context = sum_t alpha[t] * a[t] ---
    {
      const int r = tid >> 4, j0 = (tid & 15) * 4;
#pragma unroll
      for (int k = 0; k < 4; ++k) {
        const int j = j0 + k;
        float acc = 0.0f;
        for (int t = 0; t < TX; ++t)
          acc += en[r * 32 + t] * (float)aT[(t * BM + r) * ALDH + j];
        ctx16[r * CLDH + j] = (_Float16)acc;
      }
    }
    __syncthreads();

    // --- post-attn LSTM gates: z2 = ctx@W_ih_p^T + s@W_hh_p^T (N=256, 2 n-tiles/wave) ---
#pragma unroll
    for (int jj = 0; jj < 2; ++jj) {
      v8f acc = {};
#pragma unroll
      for (int c = 0; c < 2; ++c) {
        acc = wmmaf(ldsA(ctx16 + 32 * c, CLDH, lane), Bip[jj][c], acc);
        acc = wmmaf(ldsA(s16   + 32 * c, SLDH, lane), Bhp[jj][c], acc);
      }
      storeD(z2, ZLDD, 32 * wv + 16 * jj, lane, acc);
    }
    __syncthreads();

    // --- gate nonlinearities: thread -> (row, 4 hidden units of 64) ---
    {
      const int r = tid >> 4, j0 = (tid & 15) * 4;
#pragma unroll
      for (int k = 0; k < 4; ++k) {
        const int j = j0 + k;
        const float zi = z2[r * ZLDD + j]        + bP[j];
        const float zf = z2[r * ZLDD + 64 + j]   + bP[64 + j];
        const float zg = z2[r * ZLDD + 128 + j]  + bP[128 + j];
        const float zo = z2[r * ZLDD + 192 + j]  + bP[192 + j];
        cD[k] = sigf(zf) * cD[k] + sigf(zi) * ftanh(zg);
        const float s = sigf(zo) * ftanh(cD[k]);
        s16[r * SLDH + j] = (_Float16)s;
      }
    }
    __syncthreads();

    // --- output projection: out = s @ Wo^T + bo (N=64 on waves 0..3) ---
    if (wv < 4) {
      v8f acc = {};
      acc = wmmaf(ldsA(s16 + 0,  SLDH, lane), BoF[0], acc);
      acc = wmmaf(ldsA(s16 + 32, SLDH, lane), BoF[1], acc);
      const int n = lane & 15, mb = (lane >> 4) << 3, n0 = 16 * wv;
#pragma unroll
      for (int v = 0; v < 8; ++v) {
        const int row = v + mb;
        if (b0 + row < Bn)
          out[((size_t)(b0 + row) * TY + ty) * MV + n0 + n] = acc[v] + boS[n0 + n];
      }
    }
  }
}

extern "C" void kernel_launch(void* const* d_in, const int* in_sizes, int n_in,
                              void* d_out, int out_size, void* d_ws, size_t ws_size,
                              hipStream_t stream) {
  (void)n_in; (void)out_size; (void)d_ws; (void)ws_size;
  const float* X      = (const float*)d_in[0];
  const float* W_ih_f = (const float*)d_in[1];
  const float* W_hh_f = (const float*)d_in[2];
  const float* b_f    = (const float*)d_in[3];
  const float* W_ih_r = (const float*)d_in[4];
  const float* W_hh_r = (const float*)d_in[5];
  const float* b_r    = (const float*)d_in[6];
  const float* W1     = (const float*)d_in[7];
  const float* b1     = (const float*)d_in[8];
  const float* W2     = (const float*)d_in[9];
  const float* b2     = (const float*)d_in[10];
  const float* W_ih_p = (const float*)d_in[11];
  const float* W_hh_p = (const float*)d_in[12];
  const float* b_p    = (const float*)d_in[13];
  const float* Wo     = (const float*)d_in[14];
  const float* bo     = (const float*)d_in[15];
  float* out = (float*)d_out;

  const int Bn   = in_sizes[0] / (TX * HV);
  const int grid = (Bn + BM - 1) / BM;
  nmt_attn_fused<<<grid, NTH, SMEM_BYTES, stream>>>(
      X, W_ih_f, W_hh_f, b_f, W_ih_r, W_hh_r, b_r,
      W1, b1, W2, b2, W_ih_p, W_hh_p, b_p, Wo, bo, out, Bn);
}
